// MUSEPool_90872918049520
// MI455X (gfx1250) — compile-verified, matrix operand order checked by reading, not run on hardware
//
#include <hip/hip_runtime.h>
#include <hip/hip_bf16.h>
#include <math.h>

#define NN 46080
#define BB 64
#define BLKSZ 720
#define EE 400000
#define CXD 128
#define CED 32
#define CC_ITERS 40
#define SORTN 524288   // 2^19 >= EE

static const long long BIGKEY = (long long)NN * (long long)NN;

typedef __attribute__((ext_vector_type(16))) _Float16 v16h;
typedef __attribute__((ext_vector_type(8)))  float    v8f;

// ---------------- WMMA tile loaders (v_wmma_f32_16x16x32_f16 layouts) ----------------
// A (16x32 f16): lane holds row m=lane%16; halves h -> K = h + 8*(lane/16) + (h>=8?8:0)
__device__ __forceinline__ v16h load_a1(const float* __restrict__ p, int g) {
  float4 f0 = *(const float4*)(p + 8 * g);
  float4 f1 = *(const float4*)(p + 8 * g + 4);
  float4 f2 = *(const float4*)(p + 8 * g + 16);
  float4 f3 = *(const float4*)(p + 8 * g + 20);
  v16h a;
  a[0]=(_Float16)f0.x; a[1]=(_Float16)f0.y; a[2]=(_Float16)f0.z; a[3]=(_Float16)f0.w;
  a[4]=(_Float16)f1.x; a[5]=(_Float16)f1.y; a[6]=(_Float16)f1.z; a[7]=(_Float16)f1.w;
  a[8]=(_Float16)f2.x; a[9]=(_Float16)f2.y; a[10]=(_Float16)f2.z; a[11]=(_Float16)f2.w;
  a[12]=(_Float16)f3.x; a[13]=(_Float16)f3.y; a[14]=(_Float16)f3.z; a[15]=(_Float16)f3.w;
  return a;
}
// fused A = r1 + r2 (gather-sum folded into the tile load)
__device__ __forceinline__ v16h load_a2(const float* __restrict__ p1,
                                        const float* __restrict__ p2, int g) {
  float4 a0 = *(const float4*)(p1 + 8 * g);
  float4 a1 = *(const float4*)(p1 + 8 * g + 4);
  float4 a2 = *(const float4*)(p1 + 8 * g + 16);
  float4 a3 = *(const float4*)(p1 + 8 * g + 20);
  float4 b0 = *(const float4*)(p2 + 8 * g);
  float4 b1 = *(const float4*)(p2 + 8 * g + 4);
  float4 b2 = *(const float4*)(p2 + 8 * g + 16);
  float4 b3 = *(const float4*)(p2 + 8 * g + 20);
  v16h a;
  a[0]=(_Float16)(a0.x+b0.x); a[1]=(_Float16)(a0.y+b0.y); a[2]=(_Float16)(a0.z+b0.z); a[3]=(_Float16)(a0.w+b0.w);
  a[4]=(_Float16)(a1.x+b1.x); a[5]=(_Float16)(a1.y+b1.y); a[6]=(_Float16)(a1.z+b1.z); a[7]=(_Float16)(a1.w+b1.w);
  a[8]=(_Float16)(a2.x+b2.x); a[9]=(_Float16)(a2.y+b2.y); a[10]=(_Float16)(a2.z+b2.z); a[11]=(_Float16)(a2.w+b2.w);
  a[12]=(_Float16)(a3.x+b3.x); a[13]=(_Float16)(a3.y+b3.y); a[14]=(_Float16)(a3.z+b3.z); a[15]=(_Float16)(a3.w+b3.w);
  return a;
}
// B (32x16 f16, row-major W[K][ldn]): lane holds column n; same K striping as A
__device__ __forceinline__ v16h load_b(const float* __restrict__ W, int ldn,
                                       int k0, int g, int n) {
  v16h b;
#pragma unroll
  for (int h = 0; h < 16; ++h) {
    int k = k0 + ((h < 8) ? (h + 8 * g) : (h + 8 + 8 * g));
    b[h] = (_Float16)W[(size_t)k * ldn + n];
  }
  return b;
}

// ---------------- simple elementwise / scatter kernels ----------------
__global__ void k_colsum(const int* __restrict__ col, const float* __restrict__ eattr,
                         float* __restrict__ t) {
  size_t idx = (size_t)blockIdx.x * blockDim.x + threadIdx.x;
  if (idx >= (size_t)EE * CED) return;
  int e = (int)(idx >> 5), c = (int)(idx & 31);
  atomicAdd(&t[(size_t)col[e] * CED + c], eattr[idx]);
}

// ea2 = (t[row] + edge_attr) @ We + be     [WMMA, K=32, N=32]
__global__ __launch_bounds__(256) void k_edge_mlp(
    const float* __restrict__ t, const float* __restrict__ eattr,
    const int* __restrict__ row, const float* __restrict__ We,
    const float* __restrict__ be, float* __restrict__ ea2) {
  int gw = (blockIdx.x * blockDim.x + threadIdx.x) >> 5;
  int lane = threadIdx.x & 31;
  int tN = gw & 1, tM = gw >> 1;
  int g = lane >> 4, ln = lane & 15;
  int e = tM * 16 + ln;
  int r = row[e];
  v16h a = load_a2(t + (size_t)r * CED, eattr + (size_t)e * CED, g);
  int n = tN * 16 + ln;
  v16h b = load_b(We, CED, 0, g, n);
  v8f c = {};
  c = __builtin_amdgcn_wmma_f32_16x16x32_f16(false, a, false, b, (short)0, c, false, false);
#pragma unroll
  for (int v = 0; v < 8; ++v) {
    int m = tM * 16 + v + 8 * g;
    ea2[(size_t)m * CED + n] = c[v] + be[n];
  }
}

// BatchNorm (training) stats over E rows, 32 channels
__global__ void k_bnstat(const float* __restrict__ ea2, float* __restrict__ stat) {
  __shared__ float ssum[CED], ssq[CED];
  if (threadIdx.x < CED) { ssum[threadIdx.x] = 0.f; ssq[threadIdx.x] = 0.f; }
  __syncthreads();
  int c = threadIdx.x & 31;
  float ls = 0.f, lq = 0.f;
  for (size_t idx = (size_t)blockIdx.x * blockDim.x + threadIdx.x;
       idx < (size_t)EE * CED; idx += (size_t)gridDim.x * blockDim.x) {
    float v = ea2[idx];
    ls += v; lq += v * v;
  }
  atomicAdd(&ssum[c], ls); atomicAdd(&ssq[c], lq);
  __syncthreads();
  if (threadIdx.x < CED) {
    atomicAdd(&stat[threadIdx.x], ssum[threadIdx.x]);
    atomicAdd(&stat[CED + threadIdx.x], ssq[threadIdx.x]);
  }
}
__global__ void k_bnfin(float* __restrict__ stat, const float* __restrict__ gma,
                        const float* __restrict__ bta) {
  int c = threadIdx.x;
  if (c >= CED) return;
  float mean = stat[c] / (float)EE;
  float var = stat[CED + c] / (float)EE - mean * mean;
  float sc = gma[c] * rsqrtf(var + 1e-5f);
  stat[64 + c] = sc;
  stat[96 + c] = bta[c] - mean * sc;
}
__global__ void k_bnapply(float* __restrict__ ea2, const float* __restrict__ stat) {
  size_t idx = (size_t)blockIdx.x * blockDim.x + threadIdx.x;
  if (idx >= (size_t)EE * CED) return;
  int c = (int)(idx & 31);
  float v = ea2[idx] * stat[64 + c] + stat[96 + c];
  ea2[idx] = fmaxf(v, 0.f);
}

// score = sigmoid((ea2[e]+ea2[rev(e)]) . S / ||S||); group sums for mean
__global__ void k_score(const int* __restrict__ row, const int* __restrict__ col,
                        const float* __restrict__ ea2, const float* __restrict__ S,
                        const int* __restrict__ batch, float* __restrict__ score,
                        float* __restrict__ gsum, float* __restrict__ gcnt) {
  int e = blockIdx.x * blockDim.x + threadIdx.x;
  if (e >= EE) return;
  long long target = (long long)col[e] * NN + row[e];
  int lo = 0, hi = EE - 1, pos = e;
  while (lo <= hi) {
    int mid = (lo + hi) >> 1;
    long long k = (long long)row[mid] * NN + col[mid];
    if (k == target) { pos = mid; break; }
    if (k < target) lo = mid + 1; else hi = mid - 1;
  }
  float ns = 0.f, dot = 0.f;
#pragma unroll
  for (int c = 0; c < CED; ++c) {
    float sc = S[c];
    ns += sc * sc;
    dot += (ea2[(size_t)e * CED + c] + ea2[(size_t)pos * CED + c]) * sc;
  }
  float s = 1.f / (1.f + expf(-dot * rsqrtf(ns)));
  score[e] = s;
  int b = batch[row[e]];
  atomicAdd(&gsum[b], s);
  atomicAdd(&gcnt[b], 1.f);
}
__global__ void k_sel(const int* __restrict__ row, const int* __restrict__ batch,
                      const float* __restrict__ score, const float* __restrict__ gsum,
                      const float* __restrict__ gcnt, int* __restrict__ sel,
                      int* __restrict__ xmask) {
  int e = blockIdx.x * blockDim.x + threadIdx.x;
  if (e >= EE) return;
  int b = batch[row[e]];
  float gm = gsum[b] / gcnt[b];
  float s = score[e];
  int sl = (s <= 0.5f && s < gm) ? 1 : 0;
  sel[e] = sl;
  if (sl) atomicOr(&xmask[row[e]], 1);
}
__global__ void k_scale(float* __restrict__ ea2, const float* __restrict__ score) {
  size_t idx = (size_t)blockIdx.x * blockDim.x + threadIdx.x;
  if (idx >= (size_t)EE * CED) return;
  ea2[idx] *= score[idx >> 5];
}

// ---------------- connected components ----------------
__global__ void k_iota(int* __restrict__ a, int n) {
  int i = blockIdx.x * blockDim.x + threadIdx.x;
  if (i < n) a[i] = i;
}
__global__ void k_copy(int* __restrict__ d, const int* __restrict__ s, int n) {
  int i = blockIdx.x * blockDim.x + threadIdx.x;
  if (i < n) d[i] = s[i];
}
__global__ void k_ccedge(const int* __restrict__ row, const int* __restrict__ col,
                         const int* __restrict__ sel, const int* __restrict__ lab,
                         int* __restrict__ m) {
  int e = blockIdx.x * blockDim.x + threadIdx.x;
  if (e >= EE) return;
  int r = sel[e] ? row[e] : 0;
  int c = sel[e] ? col[e] : 0;
  atomicMin(&m[r], lab[c]);
  atomicMin(&m[c], lab[r]);
}
__global__ void k_jump(int* __restrict__ dst, const int* __restrict__ src, int n) {
  int i = blockIdx.x * blockDim.x + threadIdx.x;
  if (i < n) dst[i] = src[src[i]];
}
__global__ void k_isrep(const int* __restrict__ lab, int* __restrict__ o, int n) {
  int i = blockIdx.x * blockDim.x + threadIdx.x;
  if (i < n) o[i] = (lab[i] == i) ? 1 : 0;
}
__global__ void k_comp(const int* __restrict__ lab, const int* __restrict__ cum,
                       int* __restrict__ comp, int n) {
  int i = blockIdx.x * blockDim.x + threadIdx.x;
  if (i < n) comp[i] = cum[lab[i]] - 1;
}
__global__ void k_cnt(const int* __restrict__ comp, const int* __restrict__ batch,
                      int* __restrict__ cntc, int* __restrict__ bsum) {
  int i = blockIdx.x * blockDim.x + threadIdx.x;
  if (i >= NN) return;
  atomicAdd(&cntc[comp[i]], 1);
  atomicAdd(&bsum[comp[i]], batch[i]);
}
__global__ void k_bout(const int* __restrict__ cntc, const int* __restrict__ bsum,
                       int* __restrict__ out_batch) {
  int i = blockIdx.x * blockDim.x + threadIdx.x;
  if (i >= NN) return;
  int c = cntc[i];
  out_batch[i] = bsum[i] / (c > 0 ? c : 1);
}

// ---------------- 2-level inclusive scan ----------------
__global__ void k_scan_block(const int* __restrict__ in, int* __restrict__ out,
                             int* __restrict__ bsums, int n) {
  __shared__ int sh[256];
  int i = blockIdx.x * 256 + threadIdx.x;
  int v = (i < n) ? in[i] : 0;
  sh[threadIdx.x] = v;
  __syncthreads();
  for (int off = 1; off < 256; off <<= 1) {
    int t = (threadIdx.x >= off) ? sh[threadIdx.x - off] : 0;
    __syncthreads();
    sh[threadIdx.x] += t;
    __syncthreads();
  }
  if (i < n) out[i] = sh[threadIdx.x];
  if (threadIdx.x == 255) bsums[blockIdx.x] = sh[255];
}
__global__ void k_scan_sums(int* __restrict__ bsums, int nb) {
  if (blockIdx.x == 0 && threadIdx.x == 0) {
    int acc = 0;
    for (int i = 0; i < nb; ++i) { int t = bsums[i]; bsums[i] = acc; acc += t; }
  }
}
__global__ void k_scan_add(int* __restrict__ out, const int* __restrict__ bsums, int n) {
  int i = blockIdx.x * 256 + threadIdx.x;
  if (i < n) out[i] += bsums[blockIdx.x];
}

// ---------------- message GEMM: relu(x[col] + ea@W_edge), masked segment-sum ----------------
__global__ __launch_bounds__(256) void k_msg(
    const float* __restrict__ ea2, const float* __restrict__ Wedge,
    const float* __restrict__ x, const int* __restrict__ row,
    const int* __restrict__ col, const int* __restrict__ sel,
    float* __restrict__ agg) {
  int gw = (blockIdx.x * blockDim.x + threadIdx.x) >> 5;
  int lane = threadIdx.x & 31;
  int tN = gw & 7, tM = gw >> 3;
  int g = lane >> 4, ln = lane & 15;
  int n = tN * 16 + ln;
  v16h a = load_a1(ea2 + (size_t)(tM * 16 + ln) * CED, g);
  v16h b = load_b(Wedge, CXD, 0, g, n);
  v8f c = {};
  c = __builtin_amdgcn_wmma_f32_16x16x32_f16(false, a, false, b, (short)0, c, false, false);
#pragma unroll
  for (int v = 0; v < 8; ++v) {
    int e = tM * 16 + v + 8 * g;
    float val = c[v] + x[(size_t)col[e] * CXD + n];
    val = fmaxf(val, 0.f);
    if (sel[e]) atomicAdd(&agg[(size_t)row[e] * CXD + n], val);
  }
}
__global__ void k_h(const float* __restrict__ x, const float* __restrict__ agg,
                    const float* __restrict__ eps, float* __restrict__ h) {
  size_t idx = (size_t)blockIdx.x * blockDim.x + threadIdx.x;
  if (idx >= (size_t)NN * CXD) return;
  h[idx] = (1.f + eps[0]) * x[idx] + agg[idx];
}

// node GEMM: out = [relu](A @ W + bias), M=NN, K=N=128 (4 chained WMMAs)
__global__ __launch_bounds__(256) void k_gemm_node(
    const float* __restrict__ A, const float* __restrict__ W,
    const float* __restrict__ bias, float* __restrict__ out, int relu) {
  int gw = (blockIdx.x * blockDim.x + threadIdx.x) >> 5;
  int lane = threadIdx.x & 31;
  int tN = gw & 7, tM = gw >> 3;
  int g = lane >> 4, ln = lane & 15;
  const float* arow = A + (size_t)(tM * 16 + ln) * CXD;
  int n = tN * 16 + ln;
  v8f c = {};
#pragma unroll
  for (int kk = 0; kk < 4; ++kk) {
    v16h a = load_a1(arow + kk * 32, g);
    v16h b = load_b(W, CXD, kk * 32, g, n);
    c = __builtin_amdgcn_wmma_f32_16x16x32_f16(false, a, false, b, (short)0, c, false, false);
  }
#pragma unroll
  for (int v = 0; v < 8; ++v) {
    int m = tM * 16 + v + 8 * g;
    float val = c[v] + bias[n];
    if (relu) val = fmaxf(val, 0.f);
    out[(size_t)m * CXD + n] = val;
  }
}

// ---------------- pooling ----------------
__global__ void k_pool(const int* __restrict__ comp, const int* __restrict__ xmask,
                       const float* __restrict__ xconv, const float* __restrict__ x,
                       float* __restrict__ pooled, float* __restrict__ pres) {
  size_t idx = (size_t)blockIdx.x * blockDim.x + threadIdx.x;
  if (idx >= (size_t)NN * CXD) return;
  int i = (int)(idx >> 7), c = (int)(idx & 127);
  size_t o = (size_t)comp[i] * CXD + c;
  if (xmask[i]) atomicAdd(&pooled[o], xconv[idx]);
  else          atomicAdd(&pres[o], x[idx]);
}
__global__ void k_haskeep(const int* __restrict__ comp, const int* __restrict__ xmask,
                          int* __restrict__ haskeep) {
  int i = blockIdx.x * blockDim.x + threadIdx.x;
  if (i >= NN) return;
  if (!xmask[i]) atomicAdd(&haskeep[comp[i]], 1);
}
__global__ void k_xout(const int* __restrict__ haskeep, const float* __restrict__ pres,
                       const float* __restrict__ pooled, float* __restrict__ out_x) {
  size_t idx = (size_t)blockIdx.x * blockDim.x + threadIdx.x;
  if (idx >= (size_t)NN * CXD) return;
  int i = (int)(idx >> 7);
  out_x[idx] = (haskeep[i] > 0) ? pres[idx] : pooled[idx];
}
__global__ void k_xg_acc(const float* __restrict__ out_x, const int* __restrict__ out_batch,
                         float* __restrict__ xg) {
  size_t idx = (size_t)blockIdx.x * blockDim.x + threadIdx.x;
  if (idx >= (size_t)NN * CXD) return;
  int i = (int)(idx >> 7), c = (int)(idx & 127);
  atomicAdd(&xg[(size_t)out_batch[i] * CXD + c], out_x[idx]);
}
__global__ void k_xg_bn(const float* __restrict__ xg, const float* __restrict__ gma,
                        const float* __restrict__ bta, float* __restrict__ out_xg) {
  int c = threadIdx.x;
  if (c >= CXD) return;
  float mu = 0.f;
  for (int i = 0; i < BB; ++i) mu += xg[(size_t)i * CXD + c];
  mu /= (float)BB;
  float var = 0.f;
  for (int i = 0; i < BB; ++i) { float d = xg[(size_t)i * CXD + c] - mu; var += d * d; }
  var /= (float)BB;
  float inv = rsqrtf(var + 1e-5f);
  for (int i = 0; i < BB; ++i)
    out_xg[(size_t)i * CXD + c] = (xg[(size_t)i * CXD + c] - mu) * inv * gma[c] + bta[c];
}

// ---------------- edge coalesce (sort + segmented reduce) ----------------
__global__ void k_fillneg1(int* __restrict__ a, int n) {
  int i = blockIdx.x * blockDim.x + threadIdx.x;
  if (i < n) a[i] = -1;
}
__global__ void k_ckey(const int* __restrict__ row, const int* __restrict__ col,
                       const int* __restrict__ comp, const int* __restrict__ sel,
                       long long* __restrict__ sk, int* __restrict__ sv) {
  int i = blockIdx.x * blockDim.x + threadIdx.x;
  if (i >= SORTN) return;
  long long k = BIGKEY;
  if (i < EE && !sel[i]) k = (long long)comp[row[i]] * NN + comp[col[i]];
  sk[i] = k;
  sv[i] = i;
}
__global__ void k_bitonic(long long* __restrict__ key, int* __restrict__ val,
                          int k, int j) {
  int i = blockIdx.x * blockDim.x + threadIdx.x;
  int l = i ^ j;
  if (l > i) {
    long long ki = key[i], kl = key[l];
    bool asc = ((i & k) == 0);
    if (asc ? (ki > kl) : (ki < kl)) {
      key[i] = kl; key[l] = ki;
      int t = val[i]; val[i] = val[l]; val[l] = t;
    }
  }
}
__global__ void k_first(const long long* __restrict__ sk, int* __restrict__ first) {
  int i = blockIdx.x * blockDim.x + threadIdx.x;
  if (i >= EE) return;
  first[i] = (i == 0 || sk[i] != sk[i - 1]) ? 1 : 0;
}
__global__ void k_writeidx(const long long* __restrict__ sk, const int* __restrict__ first,
                           const int* __restrict__ segid, int* __restrict__ out_ei) {
  int i = blockIdx.x * blockDim.x + threadIdx.x;
  if (i >= EE || !first[i]) return;
  long long k = sk[i];
  if (k < BIGKEY) {
    int s = segid[i] - 1;
    out_ei[s] = (int)(k / NN);
    out_ei[EE + s] = (int)(k % NN);
  }
}
__global__ void k_attr(const long long* __restrict__ sk, const int* __restrict__ sv,
                       const int* __restrict__ segid, const float* __restrict__ ea2,
                       float* __restrict__ out_attr) {
  size_t idx = (size_t)blockIdx.x * blockDim.x + threadIdx.x;
  if (idx >= (size_t)EE * CED) return;
  int i = (int)(idx >> 5), c = (int)(idx & 31);
  long long k = sk[i];
  if (k >= BIGKEY) return;
  int e = sv[i];
  if (e >= EE) return;
  atomicAdd(&out_attr[(size_t)(segid[i] - 1) * CED + c], ea2[(size_t)e * CED + c]);
}

// ---------------- launch ----------------
extern "C" void kernel_launch(void* const* d_in, const int* in_sizes, int n_in,
                              void* d_out, int out_size, void* d_ws, size_t ws_size,
                              hipStream_t stream) {
  (void)in_sizes; (void)n_in; (void)out_size; (void)ws_size;
  const float* x      = (const float*)d_in[0];
  const int*   ei     = (const int*)d_in[1];
  const int*   row    = ei;
  const int*   col    = ei + EE;
  const float* eattr  = (const float*)d_in[2];
  const int*   batch  = (const int*)d_in[3];
  const float* S      = (const float*)d_in[4];
  const float* We     = (const float*)d_in[5];
  const float* be     = (const float*)d_in[6];
  const float* bn_e_g = (const float*)d_in[7];
  const float* bn_e_b = (const float*)d_in[8];
  const float* Wedge  = (const float*)d_in[9];
  const float* eps    = (const float*)d_in[10];
  const float* W1     = (const float*)d_in[11];
  const float* b1     = (const float*)d_in[12];
  const float* W2     = (const float*)d_in[13];
  const float* b2     = (const float*)d_in[14];
  const float* bn_g   = (const float*)d_in[15];
  const float* bn_b   = (const float*)d_in[16];

  float* out_x    = (float*)d_out;                                   // NN*CXD
  int*   out_ei   = (int*)((float*)d_out + (size_t)NN * CXD);        // 2*EE
  float* out_attr = (float*)d_out + (size_t)NN * CXD + 2 * (size_t)EE; // EE*CED
  int*   out_bat  = (int*)((float*)d_out + (size_t)NN * CXD + 2 * (size_t)EE + (size_t)EE * CED);
  float* out_xg   = (float*)d_out + (size_t)NN * CXD + 2 * (size_t)EE + (size_t)EE * CED + NN;

  // workspace layout
  char* base = (char*)d_ws;
  size_t off = 0;
  auto take = [&](size_t bytes) { size_t r = off; off += (bytes + 255) & ~(size_t)255; return r; };
  float*     t      = (float*)(base + take((size_t)NN * CED * 4));
  float*     ea2    = (float*)(base + take((size_t)EE * CED * 4));
  float*     bnstat = (float*)(base + take(128 * 4));
  float*     score  = (float*)(base + take((size_t)EE * 4));
  float*     gsum   = (float*)(base + take(BB * 4));
  float*     gcnt   = (float*)(base + take(BB * 4));
  int*       sel    = (int*)(base + take((size_t)EE * 4));
  int*       xmask  = (int*)(base + take((size_t)NN * 4));
  int*       lab    = (int*)(base + take((size_t)NN * 4));
  int*       mbuf   = (int*)(base + take((size_t)NN * 4));
  int*       m2buf  = (int*)(base + take((size_t)NN * 4));
  int*       comp   = (int*)(base + take((size_t)NN * 4));
  int*       cntc   = (int*)(base + take((size_t)NN * 4));
  int*       bsum   = (int*)(base + take((size_t)NN * 4));
  int*       haskp  = (int*)(base + take((size_t)NN * 4));
  int*       bsums  = (int*)(base + take(4096 * 4));
  float*     agg    = (float*)(base + take((size_t)NN * CXD * 4)); // reused as hidden
  float*     hbuf   = (float*)(base + take((size_t)NN * CXD * 4));
  float*     xconv  = (float*)(base + take((size_t)NN * CXD * 4));
  float*     pooled = (float*)(base + take((size_t)NN * CXD * 4));
  float*     pres   = (float*)(base + take((size_t)NN * CXD * 4));
  float*     xg     = (float*)(base + take((size_t)BB * CXD * 4));
  long long* sortk  = (long long*)(base + take((size_t)SORTN * 8));
  int*       sortv  = (int*)(base + take((size_t)SORTN * 4));
  int*       first  = (int*)(base + take((size_t)EE * 4));
  int*       segid  = (int*)(base + take((size_t)EE * 4));

  const int T = 256;
  const int gE   = (EE + T - 1) / T;             // 1563
  const int gEC  = (int)(((size_t)EE * CED) / T); // 50000 exact
  const int gN   = NN / T;                        // 180 exact
  const int gNC  = (int)(((size_t)NN * CXD) / T); // 23040 exact

  // zero-init
  hipMemsetAsync(t, 0, (size_t)NN * CED * 4, stream);
  hipMemsetAsync(bnstat, 0, 128 * 4, stream);
  hipMemsetAsync(gsum, 0, BB * 4, stream);
  hipMemsetAsync(gcnt, 0, BB * 4, stream);
  hipMemsetAsync(xmask, 0, (size_t)NN * 4, stream);
  hipMemsetAsync(cntc, 0, (size_t)NN * 4, stream);
  hipMemsetAsync(bsum, 0, (size_t)NN * 4, stream);
  hipMemsetAsync(haskp, 0, (size_t)NN * 4, stream);
  hipMemsetAsync(agg, 0, (size_t)NN * CXD * 4, stream);
  hipMemsetAsync(pooled, 0, (size_t)NN * CXD * 4, stream);
  hipMemsetAsync(pres, 0, (size_t)NN * CXD * 4, stream);
  hipMemsetAsync(xg, 0, (size_t)BB * CXD * 4, stream);
  hipMemsetAsync(out_attr, 0, (size_t)EE * CED * 4, stream);
  k_fillneg1<<<(2 * EE + T - 1) / T, T, 0, stream>>>(out_ei, 2 * EE);

  // edge pipeline
  k_colsum<<<gEC, T, 0, stream>>>(col, eattr, t);
  k_edge_mlp<<<(EE / 16) * 2 / 8, T, 0, stream>>>(t, eattr, row, We, be, ea2); // 6250 blocks
  k_bnstat<<<512, T, 0, stream>>>(ea2, bnstat);
  k_bnfin<<<1, 32, 0, stream>>>(bnstat, bn_e_g, bn_e_b);
  k_bnapply<<<gEC, T, 0, stream>>>(ea2, bnstat);
  k_score<<<gE, T, 0, stream>>>(row, col, ea2, S, batch, score, gsum, gcnt);
  k_sel<<<gE, T, 0, stream>>>(row, batch, score, gsum, gcnt, sel, xmask);
  k_scale<<<gEC, T, 0, stream>>>(ea2, score);

  // connected components: min-label propagation + pointer jumping
  k_iota<<<gN, T, 0, stream>>>(lab, NN);
  for (int it = 0; it < CC_ITERS; ++it) {
    k_copy<<<gN, T, 0, stream>>>(mbuf, lab, NN);
    k_ccedge<<<gE, T, 0, stream>>>(row, col, sel, lab, mbuf);
    k_jump<<<gN, T, 0, stream>>>(m2buf, mbuf, NN);
    k_jump<<<gN, T, 0, stream>>>(lab, m2buf, NN);
  }
  // compact labels
  k_isrep<<<gN, T, 0, stream>>>(lab, mbuf, NN);
  k_scan_block<<<gN, T, 0, stream>>>(mbuf, m2buf, bsums, NN);
  k_scan_sums<<<1, 1, 0, stream>>>(bsums, gN);
  k_scan_add<<<gN, T, 0, stream>>>(m2buf, bsums, NN);
  k_comp<<<gN, T, 0, stream>>>(lab, m2buf, comp, NN);
  k_cnt<<<gN, T, 0, stream>>>(comp, batch, cntc, bsum);
  k_bout<<<gN, T, 0, stream>>>(cntc, bsum, out_bat);

  // EGIN conv
  k_msg<<<(EE / 16) * 8 / 8, T, 0, stream>>>(ea2, Wedge, x, row, col, sel, agg); // 25000 blocks
  k_h<<<gNC, T, 0, stream>>>(x, agg, eps, hbuf);
  k_gemm_node<<<(NN / 16) * 8 / 8, T, 0, stream>>>(hbuf, W1, b1, agg, 1);   // hidden -> agg
  k_gemm_node<<<(NN / 16) * 8 / 8, T, 0, stream>>>(agg, W2, b2, xconv, 0);

  // pooled node features
  k_pool<<<gNC, T, 0, stream>>>(comp, xmask, xconv, x, pooled, pres);
  k_haskeep<<<gN, T, 0, stream>>>(comp, xmask, haskp);
  k_xout<<<gNC, T, 0, stream>>>(haskp, pres, pooled, out_x);

  // global add pool + BN
  k_xg_acc<<<gNC, T, 0, stream>>>(out_x, out_bat, xg);
  k_xg_bn<<<1, CXD, 0, stream>>>(xg, bn_g, bn_b, out_xg);

  // edge reduce + coalesce: bitonic sort on 2^19 (key, payload)
  k_ckey<<<SORTN / T, T, 0, stream>>>(row, col, comp, sel, sortk, sortv);
  for (int k = 2; k <= SORTN; k <<= 1)
    for (int j = k >> 1; j > 0; j >>= 1)
      k_bitonic<<<SORTN / T, T, 0, stream>>>(sortk, sortv, k, j);
  k_first<<<gE, T, 0, stream>>>(sortk, first);
  k_scan_block<<<gE, T, 0, stream>>>(first, segid, bsums, EE);
  k_scan_sums<<<1, 1, 0, stream>>>(bsums, gE);
  k_scan_add<<<gE, T, 0, stream>>>(segid, bsums, EE);
  k_writeidx<<<gE, T, 0, stream>>>(sortk, first, segid, out_ei);
  k_attr<<<gEC, T, 0, stream>>>(sortk, sortv, segid, ea2, out_attr);
}
